// Attention_42717744726641
// MI455X (gfx1250) — matrix operand order repaired and checked by reading, not compiled, not run on hardware
//
#include <hip/hip_runtime.h>
#include <hip/hip_bf16.h>
#include <math.h>

// ---------------------------------------------------------------------------
// MI455X (gfx1250) multi-head attention forward.
// bf16 WMMA (v_wmma_f32_16x16x32_bf16) everywhere, f32 accumulate.
// Flash-attention inner loop; Tensor Data Mover (TDM) stages operand tiles
// into LDS, double-buffered, tracked with TENSORcnt.
// ---------------------------------------------------------------------------

typedef __bf16 bf16_t;
typedef __attribute__((ext_vector_type(16))) __bf16 v16bf;
typedef __attribute__((ext_vector_type(8)))  __bf16 v8bf;
typedef __attribute__((ext_vector_type(8)))  float  v8f;
typedef __attribute__((ext_vector_type(4)))  unsigned int v4u;
typedef __attribute__((ext_vector_type(8)))  int v8i;
typedef __attribute__((ext_vector_type(4)))  int v4i;

union V16 { v16bf v; v8bf h[2]; bf16_t e[16]; };

#define WMMA_BF16(a, b, c) \
  __builtin_amdgcn_wmma_f32_16x16x32_bf16(false, (a), false, (b), (short)0, (c), false, false)

#if defined(__gfx1250__) && __has_builtin(__builtin_amdgcn_tensor_load_to_lds)
#define HAVE_TDM 1
#else
#define HAVE_TDM 0
#endif

// Problem dims (fixed by the reference)
constexpr int Bb = 2, Ls = 2048, Dd = 1024, Nh = 16, Hh = 64;
constexpr int Mrows = Bb * Ls;  // 4096

#if HAVE_TDM
// ---------------------------------------------------------------------------
// Issue a TDM 2D tile load (global, row-major) -> LDS. Descriptor layout per
// CDNA5 ISA ch. 8 (Tensor DMA Descriptor D#). data_size = 2 bytes (bf16).
// tile_d0 = row length in elements, tile_d1 = #rows,
// row_stride = tensor_dim0_stride in elements, gaddr = tile start.
// This toolchain exposes the 6-arg builtin:
//   (uint32x4 g0, int32x8 g1, int32x4 g2, int32x4 g3, int32x8 pad, int cpol)
// ---------------------------------------------------------------------------
__device__ __forceinline__ void tdm_load_2d(unsigned lds_off, const void* gaddr,
                                            unsigned tile_d0, unsigned tile_d1,
                                            unsigned tensor_d0, unsigned tensor_d1,
                                            unsigned row_stride) {
  unsigned long long ga = (unsigned long long)(size_t)gaddr;
  v4u g0;
  g0[0] = 1u;                                   // count=1, user descriptor
  g0[1] = lds_off;                              // lds_addr [63:32]
  g0[2] = (unsigned)(ga & 0xffffffffu);         // global_addr lo
  g0[3] = (unsigned)((ga >> 32) & 0x01ffffffu)  // global_addr hi (57-bit)
          | (2u << 30);                         // type = 2 ("image")
  v8i g1;
  g1[0] = (int)(1u << 16);                      // data_size=1 -> 2 bytes
  g1[1] = (int)(tensor_d0 << 16);               // tensor_dim0[15:0] @ bits 63:48
  g1[2] = (int)((tensor_d0 >> 16) & 0xffffu) | (int)((tensor_d1 & 0xffffu) << 16);
  g1[3] = (int)((tensor_d1 >> 16) & 0xffffu) | (int)((tile_d0 & 0xffffu) << 16);
  g1[4] = (int)(tile_d1 & 0xffffu);             // tile_dim1 (tile_dim2 = 0)
  g1[5] = (int)row_stride;                      // tensor_dim0_stride lo32
  g1[6] = 0;                                    // stride hi16 | dim1_stride lo
  g1[7] = 0;
  v4i z4 = {0, 0, 0, 0};
  v8i z8 = {0, 0, 0, 0, 0, 0, 0, 0};
  __builtin_amdgcn_tensor_load_to_lds(g0, g1, z4, z4, z8, 0);
}
// LDS byte offset of a shared-memory pointer (generic LDS addr keeps the
// offset in the low 32 bits per the flat-aperture mapping).
__device__ __forceinline__ unsigned lds_offset_of(const void* p) {
  return (unsigned)(size_t)p;
}
#endif

// ---------------------------------------------------------------------------
// fp32 -> bf16 elementwise convert
// ---------------------------------------------------------------------------
__global__ void k_cvt(const float* __restrict__ in, bf16_t* __restrict__ out, int n) {
  int i = blockIdx.x * blockDim.x + threadIdx.x;
  int stride = gridDim.x * blockDim.x;
  for (; i < n; i += stride) out[i] = (bf16_t)in[i];
}

// fp32 [R][C] row-major -> bf16 [C][R] (transposed): weights go column-major
// so every WMMA B-fragment is a contiguous 32B read.
__global__ void k_cvt_t(const float* __restrict__ in, bf16_t* __restrict__ out,
                        int R, int C) {
  int i = blockIdx.x * blockDim.x + threadIdx.x;
  int n = R * C;
  int stride = gridDim.x * blockDim.x;
  for (; i < n; i += stride) {
    int r = i / C, c = i % C;
    out[(size_t)c * R + r] = (bf16_t)in[i];
  }
}

// ---------------------------------------------------------------------------
// Projection GEMM: C[4096,1024] = X[4096,1024] * W ; W pre-transposed as
// WT[col][k]. Output stored head-major bf16 [b][n][l][h], scaled.
// Block = 128 threads = 4 waves; each wave computes a 16x64 strip.
// Weight tiles (64 cols x 32 k) staged into LDS by TDM, double-buffered.
// ---------------------------------------------------------------------------
__global__ void __launch_bounds__(128)
k_gemm_qkv(const bf16_t* __restrict__ X, const bf16_t* __restrict__ WT,
           bf16_t* __restrict__ outH, float scale) {
  const int lane = threadIdx.x & 31;
  const int wave = threadIdx.x >> 5;
  const int m16  = lane & 15;
  const int hf   = lane >> 4;           // which 16-lane half
  const int Mb   = blockIdx.x * 64 + wave * 16;
  const int Nb   = blockIdx.y * 64;

  v8f acc[4] = {};
#if HAVE_TDM
  __shared__ __attribute__((aligned(32))) bf16_t WS[2][64 * 32];
  if (wave == 0)
    tdm_load_2d(lds_offset_of(&WS[0][0]), WT + (size_t)Nb * Dd,
                /*tile_d0=*/32, /*tile_d1=*/64, /*tensor_d0=*/Dd, /*tensor_d1=*/Dd,
                /*row_stride=*/Dd);
  for (int k0 = 0; k0 < Dd; k0 += 32) {
    const int cur = (k0 >> 5) & 1;
    if (wave == 0) {
      if (k0 + 32 < Dd) {
        tdm_load_2d(lds_offset_of(&WS[cur ^ 1][0]), WT + (size_t)Nb * Dd + k0 + 32,
                    32, 64, Dd, Dd, Dd);
        __builtin_amdgcn_s_wait_tensorcnt(1);   // current tile landed
      } else {
        __builtin_amdgcn_s_wait_tensorcnt(0);
      }
    }
    __syncthreads();
    const bf16_t* ar = X + (size_t)(Mb + m16) * Dd + k0;
    V16 a;
    a.h[0] = *(const v8bf*)(ar + hf * 16);
    a.h[1] = *(const v8bf*)(ar + hf * 16 + 8);
#pragma unroll
    for (int nt = 0; nt < 4; ++nt) {
      v16bf bfrag = *(const v16bf*)(&WS[cur][(nt * 16 + m16) * 32 + hf * 16]);
      acc[nt] = WMMA_BF16(a.v, bfrag, acc[nt]);
    }
    __syncthreads();                            // all reads done before overwrite
  }
#else
  for (int k0 = 0; k0 < Dd; k0 += 32) {
    const bf16_t* ar = X + (size_t)(Mb + m16) * Dd + k0;
    V16 a;
    a.h[0] = *(const v8bf*)(ar + hf * 16);
    a.h[1] = *(const v8bf*)(ar + hf * 16 + 8);
#pragma unroll
    for (int nt = 0; nt < 4; ++nt) {
      const bf16_t* br = WT + (size_t)(Nb + nt * 16 + m16) * Dd + k0 + hf * 16;
      v16bf bfrag = *(const v16bf*)br;
      acc[nt] = WMMA_BF16(a.v, bfrag, acc[nt]);
    }
  }
#endif
  // D layout: lane l, VGPR r -> row (r + 8*hf), col (l&15). Store head-major.
#pragma unroll
  for (int nt = 0; nt < 4; ++nt) {
#pragma unroll
    for (int r = 0; r < 8; ++r) {
      int M = Mb + r + 8 * hf;
      int c = Nb + nt * 16 + m16;
      int b = M >> 11, l = M & (Ls - 1);
      int n = c >> 6, h = c & (Hh - 1);
      outH[(((size_t)b * Nh + n) * Ls + l) * Hh + h] = (bf16_t)(acc[nt][r] * scale);
    }
  }
}

// ---------------------------------------------------------------------------
// Flash attention over one (b, n): Q,K,V head-major [L][64] bf16.
// Block = 64 query rows (4 waves x 16). Key loop in tiles of 32.
// TDM streams K and V tiles into LDS (double-buffered, depth-2 pipeline).
// ---------------------------------------------------------------------------
__global__ void __launch_bounds__(128)
k_flash(const bf16_t* __restrict__ Qh, const bf16_t* __restrict__ Kh,
        const bf16_t* __restrict__ Vh, bf16_t* __restrict__ attn) {
  __shared__ __attribute__((aligned(32))) bf16_t VT[64 * 32];      // V^T tile [h][t]
  __shared__ __attribute__((aligned(32))) bf16_t PL[4 * 16 * 32];  // per-wave P tile
#if HAVE_TDM
  __shared__ __attribute__((aligned(32))) bf16_t KT[2][32 * 64];   // K tile [t][h]
  __shared__ __attribute__((aligned(32))) bf16_t VR[2][32 * 64];   // V tile [t][h]
#endif

  const int lane = threadIdx.x & 31;
  const int wave = threadIdx.x >> 5;
  const int m16  = lane & 15;
  const int hf   = lane >> 4;
  const int n = blockIdx.y, b = blockIdx.z;
  const size_t headoff = ((size_t)b * Nh + n) * (size_t)Ls * Hh;
  const bf16_t* qh = Qh + headoff;
  const bf16_t* kh = Kh + headoff;
  const bf16_t* vh = Vh + headoff;
  const int frow = blockIdx.x * 64 + wave * 16;

  // Q fragments for this wave's 16 rows: two K=32 chunks covering h=0..63
  V16 aq[2];
  {
    const bf16_t* qr = qh + (size_t)(frow + m16) * Hh;
#pragma unroll
    for (int c = 0; c < 2; ++c) {
      aq[c].h[0] = *(const v8bf*)(qr + c * 32 + hf * 16);
      aq[c].h[1] = *(const v8bf*)(qr + c * 32 + hf * 16 + 8);
    }
  }

  v8f acc[4] = {};                 // O accumulators, 16x64
  float mrow[8], lrow[8];          // per-row running max / sum (row = r + 8*hf)
#pragma unroll
  for (int r = 0; r < 8; ++r) { mrow[r] = -3.0e30f; lrow[r] = 0.0f; }
  bf16_t* pl = PL + wave * (16 * 32);

#if HAVE_TDM
  if (wave == 0) {
    tdm_load_2d(lds_offset_of(&KT[0][0]), kh, Hh, 32, Hh, Ls, Hh);
    tdm_load_2d(lds_offset_of(&VR[0][0]), vh, Hh, 32, Hh, Ls, Hh);
  }
#endif

  for (int t0 = 0; t0 < Ls; t0 += 32) {
#if HAVE_TDM
    const int cur = (t0 >> 5) & 1;
    if (wave == 0) {
      if (t0 + 32 < Ls) {
        tdm_load_2d(lds_offset_of(&KT[cur ^ 1][0]), kh + (size_t)(t0 + 32) * Hh,
                    Hh, 32, Hh, Ls, Hh);
        tdm_load_2d(lds_offset_of(&VR[cur ^ 1][0]), vh + (size_t)(t0 + 32) * Hh,
                    Hh, 32, Hh, Ls, Hh);
        __builtin_amdgcn_s_wait_tensorcnt(2);   // current K+V tiles landed
      } else {
        __builtin_amdgcn_s_wait_tensorcnt(0);
      }
    }
    __syncthreads();
    // ---- LDS-local transpose of V tile [32][64] -> VT [64][32] ----
    {
      int h = threadIdx.x & 63;
      int tseg = threadIdx.x >> 6;  // 0..1
      bf16_t tmp[16];
#pragma unroll
      for (int i = 0; i < 16; ++i)
        tmp[i] = VR[cur][(tseg * 16 + i) * Hh + h];
      v8bf* dst = (v8bf*)(VT + h * 32 + tseg * 16);
      dst[0] = *(v8bf*)&tmp[0];
      dst[1] = *(v8bf*)&tmp[8];
    }
    asm volatile("s_wait_dscnt 0" ::: "memory");
    __syncthreads();
#else
    {
      int h = threadIdx.x & 63;
      int tseg = threadIdx.x >> 6;
      bf16_t tmp[16];
#pragma unroll
      for (int i = 0; i < 16; ++i)
        tmp[i] = vh[(size_t)(t0 + tseg * 16 + i) * Hh + h];
      v8bf* dst = (v8bf*)(VT + h * 32 + tseg * 16);
      dst[0] = *(v8bf*)&tmp[0];
      dst[1] = *(v8bf*)&tmp[8];
    }
    __syncthreads();
#endif

    // ---- S = Q * K^T for 32 keys: two 16x16 f32 tiles, 4 WMMAs ----
    v8f S[2] = {};
#pragma unroll
    for (int tt = 0; tt < 2; ++tt) {
#if HAVE_TDM
      const bf16_t* kr = &KT[cur][(tt * 16 + m16) * Hh];
#else
      const bf16_t* kr = kh + (size_t)(t0 + tt * 16 + m16) * Hh;
#endif
#pragma unroll
      for (int c = 0; c < 2; ++c) {
        v16bf bk = *(const v16bf*)(kr + c * 32 + hf * 16);
        S[tt] = WMMA_BF16(aq[c].v, bk, S[tt]);
      }
    }

    // ---- online softmax (row lives across a 16-lane half) ----
#pragma unroll
    for (int r = 0; r < 8; ++r) {
      float v = fmaxf(S[0][r], S[1][r]);
      v = fmaxf(v, __shfl_xor(v, 1, 32));
      v = fmaxf(v, __shfl_xor(v, 2, 32));
      v = fmaxf(v, __shfl_xor(v, 4, 32));
      v = fmaxf(v, __shfl_xor(v, 8, 32));
      float mnew  = fmaxf(mrow[r], v);
      float alpha = __expf(mrow[r] - mnew);
      mrow[r] = mnew;
      float p0 = __expf(S[0][r] - mnew);
      float p1 = __expf(S[1][r] - mnew);
      float rs = p0 + p1;
      rs += __shfl_xor(rs, 1, 32);
      rs += __shfl_xor(rs, 2, 32);
      rs += __shfl_xor(rs, 4, 32);
      rs += __shfl_xor(rs, 8, 32);
      lrow[r] = lrow[r] * alpha + rs;
#pragma unroll
      for (int nt = 0; nt < 4; ++nt) acc[nt][r] *= alpha;
      // P (D-layout) -> LDS row-major [16][32] bf16 for A-fragment restripe
      int row = r + 8 * hf;
      pl[row * 32 + m16]      = (bf16_t)p0;
      pl[row * 32 + 16 + m16] = (bf16_t)p1;
    }
    asm volatile("s_wait_dscnt 0" ::: "memory");

    // ---- P as A-fragment (K=32 contraction over t) ----
    V16 ap;
    ap.h[0] = *(const v8bf*)(pl + m16 * 32 + hf * 16);
    ap.h[1] = *(const v8bf*)(pl + m16 * 32 + hf * 16 + 8);

    // ---- O += P * V (B-fragments from transposed V in LDS) ----
#pragma unroll
    for (int nt = 0; nt < 4; ++nt) {
      v16bf bv = *(const v16bf*)(VT + (nt * 16 + m16) * 32 + hf * 16);
      acc[nt] = WMMA_BF16(ap.v, bv, acc[nt]);
    }

#if !HAVE_TDM
    if (t0 + 32 < Ls) {
      __builtin_prefetch(kh + (size_t)(t0 + 32 + m16) * Hh, 0, 1);
      __builtin_prefetch(vh + (size_t)(t0 + 32 + m16) * Hh, 0, 1);
    }
#endif
    __syncthreads();
  }

  // ---- finalize: O / l, store f-major bf16 [M][n*64+h] ----
#pragma unroll
  for (int r = 0; r < 8; ++r) {
    float inv = 1.0f / lrow[r];
    int M = b * Ls + frow + r + 8 * hf;
#pragma unroll
    for (int nt = 0; nt < 4; ++nt)
      attn[(size_t)M * Dd + n * Hh + nt * 16 + m16] = (bf16_t)(acc[nt][r] * inv);
  }
}

// ---------------------------------------------------------------------------
// Output GEMM: out_f32[4096,1024] = attn_bf16[4096,1024] * WoT (pre-transposed)
// ---------------------------------------------------------------------------
__global__ void __launch_bounds__(128)
k_gemm_out(const bf16_t* __restrict__ A, const bf16_t* __restrict__ WT,
           float* __restrict__ out) {
  const int lane = threadIdx.x & 31;
  const int wave = threadIdx.x >> 5;
  const int m16  = lane & 15;
  const int hf   = lane >> 4;
  const int Mb   = blockIdx.x * 64 + wave * 16;
  const int Nb   = blockIdx.y * 64;

  v8f acc[4] = {};
#if HAVE_TDM
  __shared__ __attribute__((aligned(32))) bf16_t WS[2][64 * 32];
  if (wave == 0)
    tdm_load_2d(lds_offset_of(&WS[0][0]), WT + (size_t)Nb * Dd, 32, 64, Dd, Dd, Dd);
  for (int k0 = 0; k0 < Dd; k0 += 32) {
    const int cur = (k0 >> 5) & 1;
    if (wave == 0) {
      if (k0 + 32 < Dd) {
        tdm_load_2d(lds_offset_of(&WS[cur ^ 1][0]), WT + (size_t)Nb * Dd + k0 + 32,
                    32, 64, Dd, Dd, Dd);
        __builtin_amdgcn_s_wait_tensorcnt(1);
      } else {
        __builtin_amdgcn_s_wait_tensorcnt(0);
      }
    }
    __syncthreads();
    const bf16_t* ar = A + (size_t)(Mb + m16) * Dd + k0;
    V16 a;
    a.h[0] = *(const v8bf*)(ar + hf * 16);
    a.h[1] = *(const v8bf*)(ar + hf * 16 + 8);
#pragma unroll
    for (int nt = 0; nt < 4; ++nt) {
      v16bf bfrag = *(const v16bf*)(&WS[cur][(nt * 16 + m16) * 32 + hf * 16]);
      acc[nt] = WMMA_BF16(a.v, bfrag, acc[nt]);
    }
    __syncthreads();
  }
#else
  for (int k0 = 0; k0 < Dd; k0 += 32) {
    const bf16_t* ar = A + (size_t)(Mb + m16) * Dd + k0;
    V16 a;
    a.h[0] = *(const v8bf*)(ar + hf * 16);
    a.h[1] = *(const v8bf*)(ar + hf * 16 + 8);
#pragma unroll
    for (int nt = 0; nt < 4; ++nt) {
      const bf16_t* br = WT + (size_t)(Nb + nt * 16 + m16) * Dd + k0 + hf * 16;
      v16bf bfrag = *(const v16bf*)br;
      acc[nt] = WMMA_BF16(a.v, bfrag, acc[nt]);
    }
  }
#endif
#pragma unroll
  for (int nt = 0; nt < 4; ++nt) {
#pragma unroll
    for (int r = 0; r < 8; ++r) {
      int M = Mb + r + 8 * hf;
      out[(size_t)M * Dd + Nb + nt * 16 + m16] = acc[nt][r];
    }
  }
}

// ---------------------------------------------------------------------------
// Host-side launcher
// ---------------------------------------------------------------------------
extern "C" void kernel_launch(void* const* d_in, const int* in_sizes, int n_in,
                              void* d_out, int out_size, void* d_ws, size_t ws_size,
                              hipStream_t stream) {
  const float* q_in = (const float*)d_in[0];
  const float* k_in = (const float*)d_in[1];
  const float* v_in = (const float*)d_in[2];
  const float* Wq   = (const float*)d_in[3];
  const float* Wk   = (const float*)d_in[4];
  const float* Wv   = (const float*)d_in[5];
  const float* Wo   = (const float*)d_in[6];

  char* ws = (char*)d_ws;
  size_t off = 0;
  auto alloc = [&](size_t bytes) -> char* {
    char* p = ws + off;
    off += (bytes + 255) & ~(size_t)255;
    return p;
  };

  const size_t XE = (size_t)Mrows * Dd;  // 4M elements
  const size_t WE = (size_t)Dd * Dd;     // 1M elements
  bf16_t* Xq  = (bf16_t*)alloc(XE * 2);
  bf16_t* Xk  = (bf16_t*)alloc(XE * 2);
  bf16_t* Xv  = (bf16_t*)alloc(XE * 2);
  bf16_t* WqT = (bf16_t*)alloc(WE * 2);
  bf16_t* WkT = (bf16_t*)alloc(WE * 2);
  bf16_t* WvT = (bf16_t*)alloc(WE * 2);
  bf16_t* WoT = (bf16_t*)alloc(WE * 2);
  bf16_t* Qh  = (bf16_t*)alloc(XE * 2);  // [b][n][l][h]
  bf16_t* Kh  = (bf16_t*)alloc(XE * 2);
  bf16_t* Vh  = (bf16_t*)alloc(XE * 2);
  bf16_t* At  = (bf16_t*)alloc(XE * 2);  // attn, f-major [4096][1024]

  const int cvtThreads = 256;
  // Activations -> bf16 (row-major)
  k_cvt<<<dim3((int)(XE / cvtThreads)), cvtThreads, 0, stream>>>(q_in, Xq, (int)XE);
  k_cvt<<<dim3((int)(XE / cvtThreads)), cvtThreads, 0, stream>>>(k_in, Xk, (int)XE);
  k_cvt<<<dim3((int)(XE / cvtThreads)), cvtThreads, 0, stream>>>(v_in, Xv, (int)XE);
  // Weights -> bf16, transposed to [col][k]
  k_cvt_t<<<dim3((int)(WE / cvtThreads)), cvtThreads, 0, stream>>>(Wq, WqT, Dd, Dd);
  k_cvt_t<<<dim3((int)(WE / cvtThreads)), cvtThreads, 0, stream>>>(Wk, WkT, Dd, Dd);
  k_cvt_t<<<dim3((int)(WE / cvtThreads)), cvtThreads, 0, stream>>>(Wv, WvT, Dd, Dd);
  k_cvt_t<<<dim3((int)(WE / cvtThreads)), cvtThreads, 0, stream>>>(Wo, WoT, Dd, Dd);

  // Projections (Q scaled by H^-1/2 = 0.125)
  dim3 gGemm(Mrows / 64, Dd / 64);  // 64 x 16 blocks
  k_gemm_qkv<<<gGemm, 128, 0, stream>>>(Xq, WqT, Qh, 0.125f);
  k_gemm_qkv<<<gGemm, 128, 0, stream>>>(Xk, WkT, Kh, 1.0f);
  k_gemm_qkv<<<gGemm, 128, 0, stream>>>(Xv, WvT, Vh, 1.0f);

  // Flash attention per (f-block, head, batch)
  k_flash<<<dim3(Ls / 64, Nh, Bb), 128, 0, stream>>>(Qh, Kh, Vh, At);

  // Output projection to fp32
  k_gemm_out<<<gGemm, 128, 0, stream>>>(At, WoT, (float*)d_out);
}
